// DGCNN_78640851189977
// MI455X (gfx1250) — matrix operand order repaired and checked by reading, not compiled, not run on hardware
//
#include <hip/hip_runtime.h>
#include <hip/hip_bf16.h>

typedef __attribute__((ext_vector_type(16))) _Float16 v16h;
typedef __attribute__((ext_vector_type(8)))  float    v8f;

#define NPT   4096
#define NB    8
#define KNN   20
#define COUT  64
#define LEAKY 0.01f

// ------------------------------------------------------------------
// kNN: one thread per point, candidate tiles staged through LDS (float4),
// register-resident sorted top-20 (unrolled bubble insertion).
// top-k of neg_dist == top-k smallest squared distance (self included).
// ------------------------------------------------------------------
template<int C>
__global__ __launch_bounds__(128) void knn_kernel(const float* __restrict__ x,
                                                  int* __restrict__ idx_out) {
  constexpr int TILE = 128;
  __shared__ __align__(16) float tile[TILE * C];
  const int b = blockIdx.y;
  const int i = blockIdx.x * blockDim.x + threadIdx.x;
  const float* xb = x + (size_t)b * NPT * C;

  float xi[C];
#pragma unroll
  for (int c = 0; c < C; ++c) xi[c] = xb[(size_t)i * C + c];

  float bd[KNN]; int bi[KNN];
#pragma unroll
  for (int t = 0; t < KNN; ++t) { bd[t] = INFINITY; bi[t] = 0; }

  for (int j0 = 0; j0 < NPT; j0 += TILE) {
    __syncthreads();
    {
      const float4* src = (const float4*)(xb + (size_t)j0 * C);
      float4* dst = (float4*)tile;
      for (int t = threadIdx.x; t < TILE * C / 4; t += blockDim.x) dst[t] = src[t];
    }
    __syncthreads();
    for (int jj = 0; jj < TILE; ++jj) {
      float d = 0.0f;
#pragma unroll
      for (int c = 0; c < C; ++c) {
        float df = xi[c] - tile[jj * C + c];
        d = fmaf(df, df, d);
      }
      if (d < bd[KNN - 1]) {           // strict <: stable w.r.t. lower index on ties
        bd[KNN - 1] = d; bi[KNN - 1] = j0 + jj;
#pragma unroll
        for (int s = KNN - 1; s > 0; --s) {
          if (bd[s] < bd[s - 1]) {
            float td = bd[s]; bd[s] = bd[s - 1]; bd[s - 1] = td;
            int   ti = bi[s]; bi[s] = bi[s - 1]; bi[s - 1] = ti;
          }
        }
      }
    }
  }
#pragma unroll
  for (int t = 0; t < KNN; ++t)
    idx_out[((size_t)b * NPT + i) * KNN + t] = bi[t];
}

// ------------------------------------------------------------------
// Edge-conv MLP via WMMA f32<-f16 16x16x32.
// LDS holds operands PRE-SWIZZLED into a per-lane fragment layout (contiguous K),
// so each lane fetches its whole v16h fragment with one aligned 32-byte
// load (2x ds_load_b128) instead of 16 scalar ds_load_u16.
//
// A (16x32 f16) lane layout: lane = (K-half group)*16 + M,
//   halfs h -> K = h + (laneHi?16:0).
// B (32x16 f16) lane layout: lane = (K>=16)*16 + N, half h -> K = h (+16).
// C/D (16x16 f32): VGPR r -> M = r + (laneHi?8:0), N = lane%16.
// ------------------------------------------------------------------
template<int CIN>
__global__ __launch_bounds__(128) void edge_mlp_kernel(
    const float* __restrict__ xin, const int* __restrict__ idx,
    const float* __restrict__ w,   const float* __restrict__ bias,
    float* __restrict__ out) {
  constexpr int CE  = 2 * CIN;            // edge-feature channels
  constexpr int KCH = (CE + 31) / 32;     // K chunks of 32
  // fragment-layout staging:  A: [mt(2)][kc][lane(32)][h(16)]  per wave
  //                           B: [kc][nt(4)][lane(32)][h(16)]  per block
  __shared__ __align__(32) _Float16 efsA[4][2 * KCH * 32 * 16];
  __shared__ __align__(32) _Float16 wsmB[KCH * 4 * 32 * 16];

  const int tid  = threadIdx.x;
  const int wv   = tid >> 5;
  const int lane = tid & 31;

  // ---- stage weights into B-fragment layout (zero-pad K >= CE) ----
  for (int t = tid; t < KCH * 32 * COUT; t += 128) {
    int kk = t >> 6, o = t & 63;                 // kk: padded K index, o: out ch
    float val = (kk < CE) ? w[kk * COUT + o] : 0.0f;
    int kc = kk >> 5, kl = kk & 31;
    int nt = o >> 4,  nn = o & 15;
    int ln = ((kl >> 4) << 4) + nn;              // lane' = (K>=16)*16 + N
    wsmB[(((kc << 2) + nt) * 32 + ln) * 16 + (kl & 15)] = (_Float16)val;
  }

  const int p = blockIdx.x * 4 + wv;
  const int b = p / NPT, n = p % NPT;

  // ---- zero A staging with vector stores (covers K pad + edges >= 20) ----
  {
    v16h hz = {};
    v16h* za = (v16h*)&efsA[wv][0];
    for (int t = lane; t < 2 * KCH * 32; t += 32) za[t] = hz;
  }

  // ---- scatter edge features into A-fragment layout ----
  const float* xi = xin + ((size_t)b * NPT + n) * CIN;
  for (int e = 0; e < KNN; ++e) {
    int j = idx[((size_t)b * NPT + n) * KNN + e];
    const float* xj = xin + ((size_t)b * NPT + j) * CIN;
    const int mt = e >> 4, m = e & 15;
    for (int c = lane; c < CIN; c += 32) {
      float fi = xi[c], fj = xj[c];
      // channel c  -> value fi ; channel CIN + c -> value fj - fi
#pragma unroll
      for (int part = 0; part < 2; ++part) {
        int ch  = part ? (CIN + c) : c;
        float v = part ? (fj - fi) : fi;
        int kc = ch >> 5, kl = ch & 31;
        int ln = ((kl >> 4) << 4) + m;           // lane' = hiHalf*16 + M
        int h  = kl & 15;                        // half index within fragment
        efsA[wv][((mt * KCH + kc) * 32 + ln) * 16 + h] = (_Float16)v;
      }
    }
  }
  __syncthreads();

  v8f zero = {0.f, 0.f, 0.f, 0.f, 0.f, 0.f, 0.f, 0.f};
  v8f acc0[4], acc1[4];
#pragma unroll
  for (int nt = 0; nt < 4; ++nt) { acc0[nt] = zero; acc1[nt] = zero; }

  const v16h* Af = (const v16h*)&efsA[wv][0];
  const v16h* Bf = (const v16h*)&wsmB[0];
#pragma unroll
  for (int kc = 0; kc < KCH; ++kc) {
    v16h a0 = Af[(0 * KCH + kc) * 32 + lane];    // edges 0..15
    v16h a1 = Af[(1 * KCH + kc) * 32 + lane];    // edges 16..31 (masked later)
#pragma unroll
    for (int nt = 0; nt < 4; ++nt) {
      v16h bm = Bf[((kc << 2) + nt) * 32 + lane];
      acc0[nt] = __builtin_amdgcn_wmma_f32_16x16x32_f16(false, a0, false, bm,
                                                        (short)0, acc0[nt], false, false);
      acc1[nt] = __builtin_amdgcn_wmma_f32_16x16x32_f16(false, a1, false, bm,
                                                        (short)0, acc1[nt], false, false);
    }
  }

  // ---- masked max over k, merge lane halves, bias + leakyrelu, store ----
  const int M  = lane & 15;
  const int mb = (lane < 16) ? 0 : 8;
#pragma unroll
  for (int nt = 0; nt < 4; ++nt) {
    float m = -INFINITY;
#pragma unroll
    for (int r = 0; r < 8; ++r) {
      if (mb + r < KNN)      m = fmaxf(m, acc0[nt][r]);
      if (16 + mb + r < KNN) m = fmaxf(m, acc1[nt][r]);
    }
    m = fmaxf(m, __shfl_xor(m, 16, 32));   // merge M halves across lane groups
    if (lane < 16) {
      float z = m + bias[nt * 16 + M];
      z = (z >= 0.0f) ? z : LEAKY * z;
      out[((size_t)b * NPT + n) * COUT + nt * 16 + M] = z;
    }
  }
}

// ------------------------------------------------------------------
// Final projection: cat([x, x1, x2, x3]) (195 ch) @ wp (195x64) + bp.
// Thread = (row, oc); wp is 50 KB -> L2 resident, reads fully coalesced.
// ------------------------------------------------------------------
__global__ __launch_bounds__(256) void proj_kernel(
    const float* __restrict__ x,  const float* __restrict__ x1,
    const float* __restrict__ x2, const float* __restrict__ x3,
    const float* __restrict__ wp, const float* __restrict__ bp,
    float* __restrict__ out) {
  const int oc      = threadIdx.x & 63;
  const int rsub    = threadIdx.x >> 6;
  const size_t row  = (size_t)blockIdx.x * 4 + rsub;
  float acc = bp[oc];
  const float* xr = x + row * 3;
#pragma unroll
  for (int c = 0; c < 3; ++c)  acc = fmaf(xr[c], wp[c * 64 + oc], acc);
  const float* r1 = x1 + row * 64;
#pragma unroll 8
  for (int c = 0; c < 64; ++c) acc = fmaf(r1[c], wp[(3 + c) * 64 + oc], acc);
  const float* r2 = x2 + row * 64;
#pragma unroll 8
  for (int c = 0; c < 64; ++c) acc = fmaf(r2[c], wp[(67 + c) * 64 + oc], acc);
  const float* r3 = x3 + row * 64;
#pragma unroll 8
  for (int c = 0; c < 64; ++c) acc = fmaf(r3[c], wp[(131 + c) * 64 + oc], acc);
  out[row * 64 + oc] = acc;
}

// ------------------------------------------------------------------
extern "C" void kernel_launch(void* const* d_in, const int* in_sizes, int n_in,
                              void* d_out, int out_size, void* d_ws, size_t ws_size,
                              hipStream_t stream) {
  (void)in_sizes; (void)n_in; (void)out_size; (void)ws_size;
  const float* x  = (const float*)d_in[0];
  const float* w1 = (const float*)d_in[1];
  const float* b1 = (const float*)d_in[2];
  const float* w2 = (const float*)d_in[3];
  const float* b2 = (const float*)d_in[4];
  const float* w3 = (const float*)d_in[5];
  const float* b3 = (const float*)d_in[6];
  const float* wp = (const float*)d_in[7];
  const float* bp = (const float*)d_in[8];
  float* out = (float*)d_out;

  char* ws = (char*)d_ws;
  int*   idxbuf = (int*)ws;                                   // 8*4096*20*4 = 2.62 MB
  float* x1 = (float*)(ws + ((size_t)4 << 20));               // 8 MB each
  float* x2 = x1 + (size_t)NB * NPT * COUT;
  float* x3 = x2 + (size_t)NB * NPT * COUT;

  dim3 kgrid(NPT / 128, NB), kblk(128);
  dim3 egrid(NB * NPT / 4),  eblk(128);

  knn_kernel<3> <<<kgrid, kblk, 0, stream>>>(x,  idxbuf);
  edge_mlp_kernel<3> <<<egrid, eblk, 0, stream>>>(x,  idxbuf, w1, b1, x1);
  knn_kernel<64><<<kgrid, kblk, 0, stream>>>(x1, idxbuf);
  edge_mlp_kernel<64><<<egrid, eblk, 0, stream>>>(x1, idxbuf, w2, b2, x2);
  knn_kernel<64><<<kgrid, kblk, 0, stream>>>(x2, idxbuf);
  edge_mlp_kernel<64><<<egrid, eblk, 0, stream>>>(x2, idxbuf, w3, b3, x3);
  proj_kernel<<<dim3(NB * NPT / 4), dim3(256), 0, stream>>>(x, x1, x2, x3, wp, bp, out);
}
